// MixtureLinear_438086664598
// MI455X (gfx1250) — matrix operand order repaired and checked by reading, not compiled, not run on hardware
//
#include <hip/hip_runtime.h>

// ---------------------------------------------------------------------------
// MixtureLinear on MI455X (gfx1250, wave32, WMMA).
//
//   out[b,o] = sum_i input[b,i] * sum_r weight[o,i,r]*coef[b,r]
//            + sum_r bias[o,r]*coef[b,r]
//
// Strategy: GEMM proj = input(256x1024) @ weight-as-(1024 x 16384), with the
// rank-16 coef contraction fused into the epilogue of each 16x16 WMMA tile.
//
// Roofline: 8.6 GFLOP vs 66 MB of mandatory HBM traffic (weight is 64 MB f32,
// read exactly once). HBM floor at 23.3 TB/s ~= 2.9 us. bf16 WMMA
// (v_wmma_f32_16x16x32_bf16, f32 accumulate) puts compute (~1.7 us at ~5
// PFLOP/s dense bf16) under the memory floor -> memory-bound, optimal. f32
// WMMA (16x16x4) would be ~8x slower per instruction -> compute-bound at ~5x
// the HBM floor, so we convert f32 -> bf16 on the fly in registers instead.
// ---------------------------------------------------------------------------

typedef __attribute__((ext_vector_type(16))) __bf16 v16bf;
typedef __attribute__((ext_vector_type(8)))  float  v8f;

#define B_DIM   256
#define IN_DIM  1024
#define OUT_DIM 1024
#define RANK    16

__global__ __launch_bounds__(256) void mixture_linear_wmma(
    const float* __restrict__ input,   // [B, IN]
    const float* __restrict__ coef,    // [B, RANK]
    const float* __restrict__ weight,  // [OUT, IN, RANK]
    const float* __restrict__ bias,    // [OUT, RANK]
    float*       __restrict__ out)     // [B, OUT]
{
  const int lane = threadIdx.x & 31;
  const int wave = threadIdx.x >> 5;
  const int hi   = lane >> 4;          // which half-wave
  const int b0   = blockIdx.y * 16;    // 16-row b tile
  const int o    = blockIdx.x * 8 + wave;  // one output column per wave

  // --- A operand addressing: 16x32 bf16 tile of `input` (contiguous K per half-wave, same as B)
  // lanes 0-15 : row M=lane,    K = 0..15
  // lanes 16-31: row M=lane-16, K = 16..31
  const int m     = lane & 15;
  const int kbase = hi ? 16 : 0;
  const float* arow = input + (b0 + m) * IN_DIM;

  // --- B operand addressing: 32x16 bf16 tile of weight[o,:,:]
  // lanes 0-15 : column n=lane,    K = 0..15
  // lanes 16-31: column n=lane-16, K = 16..31
  // weight is contiguous in r (= n), so per-k the half-wave reads 64
  // consecutive bytes: perfectly coalesced, each weight dword read once.
  const int col  = lane & 15;
  const int koff = hi ? 16 : 0;
  const float* wcol = weight + (size_t)o * (IN_DIM * RANK) + col;

  v8f acc = {};  // 16x16 f32 accumulator (rows = b, cols = r)

  for (int k0 = 0; k0 < IN_DIM; k0 += 32) {
    // ---- A: 16 consecutive K values for this lane's row, f32 -> bf16
    const float4 a0 = *(const float4*)(arow + k0 + kbase);
    const float4 a1 = *(const float4*)(arow + k0 + kbase + 4);
    const float4 a2 = *(const float4*)(arow + k0 + kbase + 8);
    const float4 a3 = *(const float4*)(arow + k0 + kbase + 12);
    const float af[16] = {a0.x, a0.y, a0.z, a0.w, a1.x, a1.y, a1.z, a1.w,
                          a2.x, a2.y, a2.z, a2.w, a3.x, a3.y, a3.z, a3.w};
    v16bf a;
#pragma unroll
    for (int i = 0; i < 16; ++i) a[i] = (__bf16)af[i];

    // ---- B: 16 K-values of this lane's column, stride RANK floats.
    // Streamed exactly once -> non-temporal loads; prefetch next K slab.
    const float* wk = wcol + (size_t)(k0 + koff) * RANK;
    __builtin_prefetch(wk + 32 * RANK, 0, 0);
    v16bf bm;
#pragma unroll
    for (int i = 0; i < 16; ++i)
      bm[i] = (__bf16)__builtin_nontemporal_load(wk + i * RANK);

    // D = A(16x32) x B(32x16) + C, f32 accumulate
    acc = __builtin_amdgcn_wmma_f32_16x16x32_bf16(
        /*neg_a=*/false, a, /*neg_b=*/false, bm,
        /*c_mod=*/(short)0, acc, /*reuse_a=*/false, /*reuse_b=*/false);
  }

  // --- Fused epilogue: out[b,o] = sum_r (C[b,r] + bias[o,r]) * coef[b,r]
  // C layout: lane = column r; VGPR j = row (j + 8*hi). The r-sum is a
  // 16-lane xor reduction inside each half-wave.
  const int rowbase = hi ? 8 : 0;
  const float bo = bias[o * RANK + col];
  float vals[8];
#pragma unroll
  for (int j = 0; j < 8; ++j) {
    float v = (acc[j] + bo) * coef[(b0 + rowbase + j) * RANK + col];
    v += __shfl_xor(v, 1, 32);
    v += __shfl_xor(v, 2, 32);
    v += __shfl_xor(v, 4, 32);
    v += __shfl_xor(v, 8, 32);
    vals[j] = v;
  }
  if ((lane & 15) == 0) {
#pragma unroll
    for (int j = 0; j < 8; ++j)
      out[(size_t)(b0 + rowbase + j) * OUT_DIM + o] = vals[j];
  }
}

extern "C" void kernel_launch(void* const* d_in, const int* in_sizes, int n_in,
                              void* d_out, int out_size, void* d_ws, size_t ws_size,
                              hipStream_t stream) {
  const float* input  = (const float*)d_in[0];  // (256, 1024)
  const float* coef   = (const float*)d_in[1];  // (256, 16)
  const float* weight = (const float*)d_in[2];  // (1024, 1024, 16)
  const float* bias   = (const float*)d_in[3];  // (1024, 16)
  float* out = (float*)d_out;                   // (256, 1024)

  dim3 grid(OUT_DIM / 8, B_DIM / 16);  // 128 x 16 workgroups
  dim3 block(256);                     // 8 waves: one output column o each
  hipLaunchKernelGGL(mixture_linear_wmma, grid, block, 0, stream,
                     input, coef, weight, bias, out);
}
